// Block_32993938768512
// MI455X (gfx1250) — matrix operand order repaired and checked by reading, not compiled, not run on hardware
//
#include <hip/hip_runtime.h>

#define Bb 4
#define Ss 1024
#define Dd 1024
#define Hh 16
#define DHh 64
#define Ff 4096
#define Ee 8
#define Tt (Bb * Ss)          // 4096 tokens
#define CAPt (Tt / Ee)        // 512 capacity per expert

typedef __attribute__((ext_vector_type(16))) __bf16 v16bf;
typedef __attribute__((ext_vector_type(8)))  __bf16 v8bf;
typedef __attribute__((ext_vector_type(8)))  float  v8f;

__device__ __forceinline__ v8f wmma_bf16(v16bf a, v16bf b, v8f c) {
  // D = A(16x32 bf16) x B(32x16 bf16) + C(16x16 f32)
  return __builtin_amdgcn_wmma_f32_16x16x32_bf16(false, a, false, b,
                                                 (short)0, c, false, false);
}
__device__ __forceinline__ v16bf cat16(v8bf lo, v8bf hi) {
  return __builtin_shufflevector(lo, hi, 0, 1, 2, 3, 4, 5, 6, 7,
                                 8, 9, 10, 11, 12, 13, 14, 15);
}

// Per-lane async copy of one 16B chunk: global -> LDS (ASYNCcnt-tracked).
__device__ __forceinline__ void async_g2l_b128(unsigned lds_off, const void* gaddr) {
  unsigned long long ga = (unsigned long long)gaddr;
  asm volatile("global_load_async_to_lds_b128 %0, %1, off"
               :: "v"(lds_off), "v"(ga) : "memory");
}
__device__ __forceinline__ void wait_async0() {
  asm volatile("s_wait_asynccnt 0x0" ::: "memory");
}

// ---------------------------------------------------------------------------
// Pre-pass: flat f32 -> bf16 conversion.
// ---------------------------------------------------------------------------
__global__ __launch_bounds__(256)
void cvt_bf16_kernel(const float* __restrict__ in, __bf16* __restrict__ out, long n) {
  const long i0 = ((long)blockIdx.x * 256 + threadIdx.x) * 4;
  if (i0 + 3 < n) {
#pragma unroll
    for (int j = 0; j < 4; ++j) out[i0 + j] = (__bf16)in[i0 + j];
  }
}

// Pre-pass: W[K,N] f32 -> Wt[N,K] bf16 (batched). 32x32 LDS tile transpose.
__global__ __launch_bounds__(256)
void transpose_cvt_kernel(const float* __restrict__ in, __bf16* __restrict__ out,
                          int K, int N, long sIn, long sOut) {
  __shared__ float tile[32][33];
  const float* inb = in + (long)blockIdx.z * sIn;
  __bf16* outb = out + (long)blockIdx.z * sOut;
  const int n0 = blockIdx.x * 32, k0 = blockIdx.y * 32;
#pragma unroll
  for (int j = 0; j < 32; j += 8)
    tile[threadIdx.y + j][threadIdx.x] =
        inb[(long)(k0 + threadIdx.y + j) * N + n0 + threadIdx.x];
  __syncthreads();
#pragma unroll
  for (int j = 0; j < 32; j += 8)
    outb[(long)(n0 + threadIdx.y + j) * K + k0 + threadIdx.x] =
        (__bf16)tile[threadIdx.x][threadIdx.y + j];
}

// ---------------------------------------------------------------------------
// Batched GEMM, all-bf16 operands: C[b] = act(A[b] @ Wt[b]^T + bias[b])
// A: [M,K] bf16 row-major, Wt: [N,K] bf16 row-major (pre-transposed weights).
// Block (128 thr = 4 waves) computes a 64M x 64N macro-tile. The 64Nx32K
// Wt tile is double-buffered in LDS (async b128 copies, padded 80B row
// stride). Each wave: 1 A strip, 4 accumulators -> cvt-free inner loop.
// act: 0=none, 1=gelu(erf). omode: 0=f32 [M,N], 1=bf16 [M,N], 2=bf16 [N,M].
// ---------------------------------------------------------------------------
#define WROW 40   // LDS row stride in bf16 elements (80B, bank-spread)
__global__ __launch_bounds__(128)
void gemm_bt_kernel(const __bf16* __restrict__ A, const __bf16* __restrict__ Wt,
                    const float* __restrict__ bias, void* __restrict__ C,
                    int M, int N, int K,
                    long sA, long sW, long sB, long sC, int act, int omode) {
  __shared__ __bf16 wt[2][64 * WROW];           // two 5KB Wt tiles (bf16)

  const int batch = blockIdx.y;
  const __bf16* Ab = A + (long)batch * sA;
  const __bf16* Wb = Wt + (long)batch * sW;
  const float* biasb = bias + (long)batch * sB;

  const int wave = threadIdx.x >> 5;
  const int lane = threadIdx.x & 31;
  const int half = lane >> 4;
  const int r    = lane & 15;

  const int tiles_n = N >> 6;
  const int tn = blockIdx.x % tiles_n;
  const int tm = blockIdx.x / tiles_n;

  const int row = tm * 64 + wave * 16 + r;      // A row held by this lane
  const int colbase = tn * 64;
  const int nsteps = K >> 5;

  // stage K-step 0 tile into buffer 0: 64 n-rows x 32 k (bf16), 256 chunks
  {
    const unsigned ldsb = (unsigned)(uintptr_t)&wt[0][0];
#pragma unroll
    for (int i = 0; i < 2; ++i) {
      const int c = i * 128 + threadIdx.x;
      const int nl = c >> 2;                    // n_local 0..63
      const int ko = (c & 3) * 8;               // k offset (elements)
      async_g2l_b128(ldsb + (unsigned)((nl * WROW + ko) * 2),
                     Wb + (long)(colbase + nl) * K + ko);
    }
  }

  v8f acc[4] = {};
  for (int it = 0; it < nsteps; ++it) {
    const int kk = it << 5;
    wait_async0();                              // my staged chunks landed
    __syncthreads();                            // everyone's chunks landed

    if (it + 1 < nsteps) {                      // prefetch next tile (overlap)
      const unsigned ldsb = (unsigned)(uintptr_t)&wt[(it + 1) & 1][0];
      const __bf16* Wn = Wb + kk + 32;
#pragma unroll
      for (int i = 0; i < 2; ++i) {
        const int c = i * 128 + threadIdx.x;
        const int nl = c >> 2;
        const int ko = (c & 3) * 8;
        async_g2l_b128(ldsb + (unsigned)((nl * WROW + ko) * 2),
                       Wn + (long)(colbase + nl) * K + ko);
      }
    }

    // A fragment: two contiguous 16B bf16 loads, no conversion
    const __bf16* ap = Ab + (long)row * K + kk;
    __builtin_prefetch(ap + 32, 0, 0);
    const v16bf a = cat16(*(const v8bf*)(ap + half * 8),
                          *(const v8bf*)(ap + 16 + half * 8));

    const __bf16* bt = &wt[it & 1][0];
#pragma unroll
    for (int g = 0; g < 4; ++g) {               // 4 col groups share the A frag
      const __bf16* bp = bt + (g * 16 + r) * WROW + half * 8;
      const v16bf b = cat16(*(const v8bf*)bp, *(const v8bf*)(bp + 16));
      acc[g] = wmma_bf16(a, b, acc[g]);
    }
  }

#pragma unroll
  for (int g = 0; g < 4; ++g) {
    const int col = colbase + g * 16 + r;
    const float bv = biasb[col];
#pragma unroll
    for (int i = 0; i < 8; ++i) {               // C row = row base + half*8 + i
      const long m = tm * 64 + wave * 16 + half * 8 + i;
      float vv = acc[g][i] + bv;
      if (act == 1) vv = vv * 0.5f * (1.0f + erff(vv * (1.0f / 1.41421f)));
      if (omode == 0)      ((float*)C  + (long)batch * sC)[m * N + col] = vv;
      else if (omode == 1) ((__bf16*)C + (long)batch * sC)[m * N + col] = (__bf16)vv;
      else                 ((__bf16*)C + (long)batch * sC)[(long)col * M + m] = (__bf16)vv;
    }
  }
}

// ---------------------------------------------------------------------------
// scores[b,h,q,k] = dot(Q[b,q,h,:], K[b,k,h,:]) * 0.125 + mask[b,k]
// Q,K are bf16 [T,D]; one wave computes 16q x 64k (Q fragment reused 4x).
// ---------------------------------------------------------------------------
__global__ __launch_bounds__(128)
void scores_kernel(const __bf16* __restrict__ q, const __bf16* __restrict__ kmat,
                   const float* __restrict__ mask, float* __restrict__ at) {
  const int bh = blockIdx.y;
  const int b = bh / Hh, h = bh % Hh;
  const int wave = threadIdx.x >> 5;
  const int lane = threadIdx.x & 31;
  const int half = lane >> 4;
  const int r    = lane & 15;

  const int tile = blockIdx.x * 4 + wave;   // (S/16) q-tiles x (S/64) k-tiles
  const int tnk = tile & 15;
  const int tmq = tile >> 4;

  const int qrow = b * Ss + tmq * 16 + r;

  v8f acc[4] = {};
#pragma unroll
  for (int kk = 0; kk < DHh; kk += 32) {
    const __bf16* ap = q + (long)qrow * Dd + h * DHh + kk;
    const v16bf a = cat16(*(const v8bf*)(ap + half * 8),
                          *(const v8bf*)(ap + 16 + half * 8));
#pragma unroll
    for (int g = 0; g < 4; ++g) {
      const int krow = b * Ss + tnk * 64 + g * 16 + r;   // key row (B column)
      const __bf16* kp = kmat + (long)krow * Dd + h * DHh + kk + half * 8;
      const v16bf bt = cat16(*(const v8bf*)kp, *(const v8bf*)(kp + 16));
      acc[g] = wmma_bf16(a, bt, acc[g]);
    }
  }

#pragma unroll
  for (int g = 0; g < 4; ++g) {
    const int kcol = tnk * 64 + g * 16 + r;
    const float mk = mask[b * Ss + kcol];
    float* out = at + ((long)bh * Ss + tmq * 16) * Ss + kcol;
#pragma unroll
    for (int i = 0; i < 8; ++i)
      out[(long)(half * 8 + i) * Ss] = acc[g][i] * 0.125f + mk;
  }
}

// ---------------------------------------------------------------------------
// Fused softmax + context: ctx[16,64] = softmax(scores[16,S]) @ V[S,64]
// V is pre-transposed bf16 [D, T] so B fragments are contiguous.
// ---------------------------------------------------------------------------
__global__ __launch_bounds__(128)
void softmax_ctx_kernel(const float* __restrict__ at, const __bf16* __restrict__ vT,
                        __bf16* __restrict__ ctx) {
  const int wave = threadIdx.x >> 5;
  const int lane = threadIdx.x & 31;
  const int half = lane >> 4;
  const int r    = lane & 15;

  const int tile = blockIdx.x * 4 + wave;  // over B*H*(S/16)
  const int qb = tile & 63;
  const int bh = tile >> 6;
  const int b = bh / Hh, h = bh % Hh;

  const float* srow = at + ((long)bh * Ss + qb * 16 + r) * Ss;

  float mx = -3.4e38f;
  for (int j = half; j < Ss; j += 2) mx = fmaxf(mx, srow[j]);
  mx = fmaxf(mx, __shfl_xor(mx, 16, 32));
  float sum = 0.f;
  for (int j = half; j < Ss; j += 2) sum += __expf(srow[j] - mx);
  sum += __shfl_xor(sum, 16, 32);
  const float rcp = 1.0f / sum;

  v8f acc[4] = {};
  for (int kk = 0; kk < Ss; kk += 32) {
    v16bf a;
    const float* sp0 = srow + kk + half * 8;
    const float* sp1 = sp0 + 16;
#pragma unroll
    for (int e = 0; e < 8; ++e) {
      a[e]     = (__bf16)(__expf(sp0[e] - mx) * rcp);
      a[e + 8] = (__bf16)(__expf(sp1[e] - mx) * rcp);
    }
#pragma unroll
    for (int g = 0; g < 4; ++g) {          // 4 groups of 16 head-dim columns
      const __bf16* vp = vT + (long)(h * DHh + g * 16 + r) * Tt
                            + b * Ss + kk + half * 8;
      const v16bf bt = cat16(*(const v8bf*)vp, *(const v8bf*)(vp + 16));
      acc[g] = wmma_bf16(a, bt, acc[g]);
    }
  }

#pragma unroll
  for (int g = 0; g < 4; ++g)
#pragma unroll
    for (int i = 0; i < 8; ++i)
      ctx[((long)(b * Ss + qb * 16 + half * 8 + i)) * Dd + h * DHh + g * 16 + r] =
          (__bf16)acc[g][i];
}

// ---------------------------------------------------------------------------
// out[row] = ln_w * ((a+res) - mean)/sqrt(var + 1e-12) + ln_b  (one block/row)
// ---------------------------------------------------------------------------
__global__ __launch_bounds__(256)
void add_ln_kernel(const float* __restrict__ a, const float* __restrict__ res,
                   const float* __restrict__ w, const float* __restrict__ bias,
                   float* __restrict__ out) {
  __shared__ float red[256];
  const int row = blockIdx.x;
  const int t = threadIdx.x;
  const float* ar = a + (long)row * Dd;
  const float* rr = res + (long)row * Dd;

  float vals[4];
  float s = 0.f;
#pragma unroll
  for (int j = 0; j < 4; ++j) {
    const int idx = t + j * 256;
    vals[j] = ar[idx] + rr[idx];
    s += vals[j];
  }
  red[t] = s; __syncthreads();
  for (int off = 128; off > 0; off >>= 1) {
    if (t < off) red[t] += red[t + off];
    __syncthreads();
  }
  const float mean = red[0] * (1.0f / Dd);
  __syncthreads();

  float s2 = 0.f;
#pragma unroll
  for (int j = 0; j < 4; ++j) { const float d = vals[j] - mean; s2 += d * d; }
  red[t] = s2; __syncthreads();
  for (int off = 128; off > 0; off >>= 1) {
    if (t < off) red[t] += red[t + off];
    __syncthreads();
  }
  const float inv = 1.0f / sqrtf(red[0] * (1.0f / Dd) + 1e-12f);

#pragma unroll
  for (int j = 0; j < 4; ++j) {
    const int idx = t + j * 256;
    out[(long)row * Dd + idx] = w[idx] * (vals[j] - mean) * inv + bias[idx];
  }
}

// ---------------------------------------------------------------------------
// Router: routes[t] = argmax(x1[t] @ Wsw + bsw). One wave per token.
// ---------------------------------------------------------------------------
__global__ __launch_bounds__(256)
void router_kernel(const float* __restrict__ x1, const float* __restrict__ Wsw,
                   const float* __restrict__ bsw, int* __restrict__ routes) {
  const int wave = threadIdx.x >> 5;
  const int lane = threadIdx.x & 31;
  const int t = blockIdx.x * 8 + wave;
  const int e = lane & 7;
  const int c = lane >> 3;

  const float* xr = x1 + (long)t * Dd;
  float s = 0.f;
  const int d0 = c * 256;
  for (int d = d0; d < d0 + 256; ++d) s += xr[d] * Wsw[d * Ee + e];
  s += __shfl_xor(s, 8, 32);
  s += __shfl_xor(s, 16, 32);
  s += bsw[e];

  int idx = e;
  for (int d = 1; d < 8; d <<= 1) {   // argmax, ties -> lowest index
    const float ov = __shfl_xor(s, d, 8);
    const int   oi = __shfl_xor(idx, d, 8);
    if (ov > s || (ov == s && oi < idx)) { s = ov; idx = oi; }
  }
  if (lane == 0) routes[t] = idx;
}

// Deterministic in-order cumulative slot assignment (tiny, order-sensitive).
__global__ void route_cumsum_kernel(const int* __restrict__ routes,
                                    int* __restrict__ slot) {
  if (threadIdx.x == 0 && blockIdx.x == 0) {
    int counts[Ee];
    for (int e = 0; e < Ee; ++e) counts[e] = 0;
    for (int t = 0; t < Tt; ++t) {
      const int rte = routes[t];
      const int p = counts[rte]++;
      slot[t] = (p < CAPt) ? rte * CAPt + p : -1;
    }
  }
}

// Scatter kept tokens into expert capacity buffer [E*cap, D] (f32 -> bf16).
__global__ __launch_bounds__(256)
void scatter_bf_kernel(const float* __restrict__ x1, const int* __restrict__ slot,
                       __bf16* __restrict__ ebuf) {
  const int t = blockIdx.x;
  const int s = slot[t];
  if (s < 0) return;
  const float* src = x1 + (long)t * Dd;
  __bf16* dst = ebuf + (long)s * Dd;
#pragma unroll
  for (int j = 0; j < 4; ++j) {
    const int i = threadIdx.x * 4 + j;
    dst[i] = (__bf16)src[i];
  }
}

// Gather expert output back to token order; dropped tokens pass through x1.
__global__ __launch_bounds__(256)
void gather_kernel(const float* __restrict__ ffnbuf, const int* __restrict__ slot,
                   const float* __restrict__ x1, float* __restrict__ out) {
  const int t = blockIdx.x;
  const int s = slot[t];
  const float4* src = (s >= 0) ? (const float4*)(ffnbuf + (long)s * Dd)
                               : (const float4*)(x1 + (long)t * Dd);
  float4* dst = (float4*)(out + (long)t * Dd);
  dst[threadIdx.x] = src[threadIdx.x];
}

// ---------------------------------------------------------------------------
extern "C" void kernel_launch(void* const* d_in, const int* in_sizes, int n_in,
                              void* d_out, int out_size, void* d_ws, size_t ws_size,
                              hipStream_t stream) {
  (void)in_sizes; (void)n_in; (void)out_size; (void)ws_size;

  const float* x    = (const float*)d_in[0];
  const float* mask = (const float*)d_in[1];
  const float* Wq = (const float*)d_in[2];  const float* bq = (const float*)d_in[3];
  const float* Wk = (const float*)d_in[4];  const float* bk = (const float*)d_in[5];
  const float* Wv = (const float*)d_in[6];  const float* bv = (const float*)d_in[7];
  const float* Wo = (const float*)d_in[8];  const float* bo = (const float*)d_in[9];
  const float* ln1w = (const float*)d_in[10]; const float* ln1b = (const float*)d_in[11];
  const float* ln2w = (const float*)d_in[12]; const float* ln2b = (const float*)d_in[13];
  const float* Wsw = (const float*)d_in[14]; const float* bsw = (const float*)d_in[15];
  const float* W1  = (const float*)d_in[16]; const float* b1  = (const float*)d_in[17];
  const float* W2  = (const float*)d_in[18]; const float* b2  = (const float*)d_in[19];

  float* out_x = (float*)d_out;                        // [B,S,D]
  float* at    = out_x + (long)Bb * Ss * Dd;           // [B,H,S,S]

  // ---- workspace layout (byte offsets) ----
  const size_t MB = 1ull << 20;
  char* wsb = (char*)d_ws;
  __bf16* W1T   = (__bf16*)(wsb + 0 * MB);      // [E][F,D]   64MB
  __bf16* W2T   = (__bf16*)(wsb + 64 * MB);     // [E][D,F]   64MB
  __bf16* WqT   = (__bf16*)(wsb + 128 * MB);    // [D,D]       2MB
  __bf16* WkT   = (__bf16*)(wsb + 130 * MB);
  __bf16* WvT   = (__bf16*)(wsb + 132 * MB);
  __bf16* WoT   = (__bf16*)(wsb + 134 * MB);
  __bf16* xbf   = (__bf16*)(wsb + 136 * MB);    // [T,D]       8MB
  __bf16* qbf   = (__bf16*)(wsb + 144 * MB);    // [T,D]       8MB
  __bf16* kbf   = (__bf16*)(wsb + 152 * MB);    // [T,D]       8MB
  __bf16* vT    = (__bf16*)(wsb + 160 * MB);    // [D,T]       8MB
  __bf16* ctxbf = (__bf16*)(wsb + 168 * MB);    // [T,D]       8MB
  float*  attn  = (float*)(wsb + 176 * MB);     // [T,D]      16MB
  float*  x1    = (float*)(wsb + 192 * MB);     // [T,D]      16MB
  __bf16* ebuf  = (__bf16*)(wsb + 208 * MB);    // [E*cap,D]   8MB
  __bf16* hbuf  = (__bf16*)(wsb + 216 * MB);    // [E*cap,F]  32MB
  float*  ffnb  = (float*)(wsb + 248 * MB);     // [E*cap,D]  16MB
  float*  ffntk = (float*)(wsb + 264 * MB);     // [T,D]      16MB
  int*    routes = (int*)(wsb + 280 * MB);
  int*    slot   = routes + Tt;

  const dim3 blk128(128), blk256(256), blkT(32, 8);

  // ---- pre-passes: bf16 conversion + weight transposition ----
  cvt_bf16_kernel<<<dim3((Tt * Dd) / 1024), blk256, 0, stream>>>(x, xbf, (long)Tt * Dd);
  transpose_cvt_kernel<<<dim3(Dd / 32, Dd / 32, 1), blkT, 0, stream>>>(Wq, WqT, Dd, Dd, 0, 0);
  transpose_cvt_kernel<<<dim3(Dd / 32, Dd / 32, 1), blkT, 0, stream>>>(Wk, WkT, Dd, Dd, 0, 0);
  transpose_cvt_kernel<<<dim3(Dd / 32, Dd / 32, 1), blkT, 0, stream>>>(Wv, WvT, Dd, Dd, 0, 0);
  transpose_cvt_kernel<<<dim3(Dd / 32, Dd / 32, 1), blkT, 0, stream>>>(Wo, WoT, Dd, Dd, 0, 0);
  transpose_cvt_kernel<<<dim3(Ff / 32, Dd / 32, Ee), blkT, 0, stream>>>(
      W1, W1T, Dd, Ff, (long)Dd * Ff, (long)Ff * Dd);
  transpose_cvt_kernel<<<dim3(Dd / 32, Ff / 32, Ee), blkT, 0, stream>>>(
      W2, W2T, Ff, Dd, (long)Ff * Dd, (long)Dd * Ff);

  // ---- attention ----
  const dim3 gQ((Tt / 64) * (Dd / 64), 1);
  gemm_bt_kernel<<<gQ, blk128, 0, stream>>>(xbf, WqT, bq, qbf, Tt, Dd, Dd, 0, 0, 0, 0, 0, 1);
  gemm_bt_kernel<<<gQ, blk128, 0, stream>>>(xbf, WkT, bk, kbf, Tt, Dd, Dd, 0, 0, 0, 0, 0, 1);
  gemm_bt_kernel<<<gQ, blk128, 0, stream>>>(xbf, WvT, bv, vT,  Tt, Dd, Dd, 0, 0, 0, 0, 0, 2);

  scores_kernel<<<dim3((Ss / 16) * (Ss / 64) / 4, Bb * Hh), blk128, 0, stream>>>(
      qbf, kbf, mask, at);
  softmax_ctx_kernel<<<dim3(Bb * Hh * (Ss / 16) / 4), blk128, 0, stream>>>(at, vT, ctxbf);

  gemm_bt_kernel<<<gQ, blk128, 0, stream>>>(ctxbf, WoT, bo, attn, Tt, Dd, Dd, 0, 0, 0, 0, 0, 0);
  add_ln_kernel<<<dim3(Tt), blk256, 0, stream>>>(attn, x, ln1w, ln1b, x1);

  // ---- switch FFN ----
  router_kernel<<<dim3(Tt / 8), blk256, 0, stream>>>(x1, Wsw, bsw, routes);
  route_cumsum_kernel<<<dim3(1), dim3(64), 0, stream>>>(routes, slot);
  scatter_bf_kernel<<<dim3(Tt), blk256, 0, stream>>>(x1, slot, ebuf);

  gemm_bt_kernel<<<dim3((CAPt / 64) * (Ff / 64), Ee), blk128, 0, stream>>>(
      ebuf, W1T, b1, hbuf, CAPt, Ff, Dd,
      (long)CAPt * Dd, (long)Ff * Dd, (long)Ff, (long)CAPt * Ff, 1, 1);
  gemm_bt_kernel<<<dim3((CAPt / 64) * (Dd / 64), Ee), blk128, 0, stream>>>(
      hbuf, W2T, b2, ffnb, CAPt, Dd, Ff,
      (long)CAPt * Ff, (long)Dd * Ff, (long)Dd, (long)CAPt * Dd, 0, 0);

  gather_kernel<<<dim3(Tt), blk256, 0, stream>>>(ffnb, slot, x1, ffntk);
  add_ln_kernel<<<dim3(Tt), blk256, 0, stream>>>(ffntk, x1, ln2w, ln2b, out_x);
}
